// SMOLEnv_65335042506807
// MI455X (gfx1250) — matrix operand order repaired and checked by reading, not compiled, not run on hardware
//
#include <hip/hip_runtime.h>

typedef _Float16 half_t;
typedef half_t v16h __attribute__((ext_vector_type(16)));
typedef float  v8f  __attribute__((ext_vector_type(8)));

#define N_ATOMS   2048
#define N_HEADS   5
#define RBF_BINS  16

// centers: linspace(0,10,16) -> step 10/15 ; sigma = 10/16 = 0.625 ; 1/sigma = 1.6
#define CENTER_STEP (10.0f / 15.0f)
#define INV_SIGMA   1.6f

__global__ __launch_bounds__(256) void smol_env_attn_kernel(
    const float* __restrict__ pos,   // [B, N, 3]
    const float* __restrict__ mask,  // [B, N]
    const float* __restrict__ W,     // [16, 5] row-major
    float* __restrict__ out,         // [B, N, 5, 3]
    int B)
{
    // {x, y, z, mask_bit} per atom: one ds_load_b128 per neighbor element.
    __shared__ float4 s_p[N_ATOMS];

    const int tid            = threadIdx.x;
    const int rows_per_block = blockDim.x >> 5;          // 8 waves -> 8 query rows
    const int blocks_per_b   = N_ATOMS / rows_per_block; // 256
    const int b              = blockIdx.x / blocks_per_b;
    const int n0             = (blockIdx.x % blocks_per_b) * rows_per_block;

    // Stage this batch's positions + binarized mask into LDS (32 KB of 320 KB/WGP).
    for (int i = tid; i < N_ATOMS; i += blockDim.x) {
        const float* p = pos + (size_t)(b * N_ATOMS + i) * 3;
        float4 v;
        v.x = p[0];
        v.y = p[1];
        v.z = p[2];
        v.w = (mask[b * N_ATOMS + i] > 0.0f) ? 1.0f : 0.0f; // pair-mask bit
        s_p[i] = v;
    }
    __syncthreads();

    const int wave = tid >> 5;
    const int lane = tid & 31;
    const int mrow = lane & 15;   // neighbor row within tile (also head column for C)
    const int half = lane >> 4;   // 0: K=0..7 / C rows 0..7 ; 1: K=8..15 / C rows 8..15
    const int n    = n0 + wave;

    const float4 q = s_p[n];      // query atom (w = row mask bit)

    // B operand: 32(K) x 16(N) f16. Lanes 0-15 hold K=0..7 (+16..23 zero pad),
    // lanes 16-31 hold K=8..15 (+24..31 zero pad). Column = head index, heads >=5 zero.
    const int h = mrow;
    v16h bmat;
    #pragma unroll
    for (int j = 0; j < 16; ++j) {
        float w = (j < 8 && h < N_HEADS) ? W[(8 * half + j) * N_HEADS + h] : 0.0f;
        bmat[j] = (half_t)w;
    }

    float sum = 0.0f, ax = 0.0f, ay = 0.0f, az = 0.0f;

    for (int tile = 0; tile < N_ATOMS; tile += 16) {
        // Distance for this lane's neighbor row (duplicated across halves).
        const float4 pm = s_p[tile + mrow];
        const float dx = q.x - pm.x;
        const float dy = q.y - pm.y;
        const float dz = q.z - pm.z;
        const float d2 = fmaxf(dx * dx + dy * dy + dz * dz, 1e-12f); // EPS_NORM^2
        const float d  = sqrtf(d2);

        // A operand: 16(M) x 32(K) f16. Lanes 0-15 hold K=0..7 (+16..23),
        // lanes 16-31 hold K=8..15 (+24..31); K>=16 zero-padded.
        v16h a;
        #pragma unroll
        for (int j = 0; j < 8; ++j) {
            const int   k = 8 * half + j;
            const float c = CENTER_STEP * (float)k;
            const float t = (d - c) * INV_SIGMA;
            a[j] = (half_t)__expf(-t * t);
        }
        #pragma unroll
        for (int j = 8; j < 16; ++j) a[j] = (half_t)0.0f;

        // logits[tile-row, head] = feat @ W  (K=16 real, padded to 32)
        v8f cacc = {};
        cacc = __builtin_amdgcn_wmma_f32_16x16x32_f16(
            /*neg_a=*/false, a, /*neg_b=*/false, bmat,
            /*c_mod=*/(short)0, cacc, /*reuse_a=*/false, /*reuse_b=*/false);

        // C layout: VGPR j -> row M=j (lanes 0-15) / M=j+8 (lanes 16-31), col N=lane%16.
        // Branch-free accumulation: mask gates multiplicatively (exp(-1e9) -> 0).
        #pragma unroll
        for (int j = 0; j < 8; ++j) {
            const int    mm = tile + j + 8 * half;
            const float4 pv = s_p[mm];
            const float  w  = __expf(cacc[j]) * pv.w; // |logit| <= sum|W|: no max-sub
            sum += w;
            ax  += w * pv.x;
            ay  += w * pv.y;
            az  += w * pv.z;
        }
    }

    // Merge the two m-halves (same head, complementary rows). wave32 shuffle.
    sum += __shfl_xor(sum, 16, 32);
    ax  += __shfl_xor(ax, 16, 32);
    ay  += __shfl_xor(ay, 16, 32);
    az  += __shfl_xor(az, 16, 32);

    if (half == 0 && h < N_HEADS) {
        // Row mask factored out of the sum: all-masked row (or mask_n==0) -> zeros.
        const float inv  = (q.w > 0.0f && sum > 0.0f) ? (1.0f / sum) : 0.0f;
        const size_t base = (((size_t)(b * N_ATOMS + n)) * N_HEADS + h) * 3;
        if (h == 1) { // reference zeroes head 1
            out[base + 0] = 0.0f;
            out[base + 1] = 0.0f;
            out[base + 2] = 0.0f;
        } else {
            out[base + 0] = ax * inv;
            out[base + 1] = ay * inv;
            out[base + 2] = az * inv;
        }
    }
}

extern "C" void kernel_launch(void* const* d_in, const int* in_sizes, int n_in,
                              void* d_out, int out_size, void* d_ws, size_t ws_size,
                              hipStream_t stream) {
    (void)n_in; (void)out_size; (void)d_ws; (void)ws_size;
    const float* pos  = (const float*)d_in[0];   // [B, 2048, 3]
    // d_in[1] = smol_atom_type (unused by the reference)
    const float* mask = (const float*)d_in[2];   // [B, 2048]
    const float* W    = (const float*)d_in[3];   // [16, 5]
    float* out        = (float*)d_out;           // [B, 2048, 5, 3]

    const int B = in_sizes[2] / N_ATOMS;         // mask has B*N elements
    const int rows_per_block = 256 / 32;         // 8 waves per block
    const int grid = B * (N_ATOMS / rows_per_block);

    smol_env_attn_kernel<<<grid, 256, 0, stream>>>(pos, mask, W, out, B);
}
